// MultiHeadAttention_87703232184396
// MI455X (gfx1250) — matrix operand order repaired and checked by reading, not compiled, not run on hardware
//
#include <hip/hip_runtime.h>
#include <hip/hip_bf16.h>
#include <math.h>

// Problem constants (from reference): B=2, S=2048, D=1024, H=16, DH=64
#define B_  2
#define S_  2048
#define D_  1024
#define H_  16
#define DH_ 64

typedef __attribute__((ext_vector_type(16))) _Float16 v16h;
typedef __attribute__((ext_vector_type(8)))  _Float16 v8h;
typedef __attribute__((ext_vector_type(8)))  float    v8f;

__device__ __forceinline__ v8f wmma16x16x32(v16h a, v16h b, v8f c) {
  // D = A(16x32 f16) * B(32x16 f16) + C(16x16 f32)
  return __builtin_amdgcn_wmma_f32_16x16x32_f16(
      /*neg_a=*/false, a, /*neg_b=*/false, b,
      /*c_mod=*/(short)0, c, /*reuse_a=*/false, /*reuse_b=*/false);
}

// A-fragment loaders (16x32 f16 tile, per-lane: rows r=lane&15, K group g).
// f16 source: two packed 16-byte loads. f32 source: load + convert.
__device__ __forceinline__ void load_afrag(const _Float16* __restrict__ Arow,
                                           int kA, v16h& a) {
  const v8h a0 = *(const v8h*)(Arow + kA);
  const v8h a1 = *(const v8h*)(Arow + kA + 8);
#pragma unroll
  for (int j = 0; j < 8; ++j) { a[j] = a0[j]; a[j + 8] = a1[j]; }
}
__device__ __forceinline__ void load_afrag(const float* __restrict__ Arow,
                                           int kA, v16h& a) {
#pragma unroll
  for (int j = 0; j < 8; ++j) {
    a[j]     = (_Float16)Arow[kA + j];
    a[j + 8] = (_Float16)Arow[kA + 8 + j];
  }
}

__device__ __forceinline__ void store_c(float* p, float v)     { *p = v; }
__device__ __forceinline__ void store_c(_Float16* p, float v)  { *p = (_Float16)v; }

// ---------------------------------------------------------------------------
// Generic C = A[MxK] @ W[KxN] + bias. A is f32 or f16; W,bias f32;
// C is f16 (workspace activations) or f32 (final output). f16 WMMA compute.
// Block = 128 threads = 4 waves; each wave owns one 16x16 output tile.
// grid = (N/64, M/16). M,N,K all multiples of 32 here, no bounds checks.
// ---------------------------------------------------------------------------
template <typename AT, typename CT>
__global__ void gemm_bias_kernel(const AT* __restrict__ A,
                                 const float* __restrict__ W,
                                 const float* __restrict__ bias,
                                 CT* __restrict__ C,
                                 int M, int N, int K) {
  const int lane   = threadIdx.x & 31;
  const int wave   = threadIdx.x >> 5;
  const int tile_n = (blockIdx.x * 4 + wave) * 16;
  const int tile_m = blockIdx.y * 16;
  const int r = lane & 15;   // A row-in-tile; B/C col-in-tile
  const int g = lane >> 4;   // lane group selects K sub-range

  const AT* Arow = A + (size_t)(tile_m + r) * K;
  v8f acc = {};
  for (int kk = 0; kk < K; kk += 32) {
    v16h a, b;
    load_afrag(Arow, kk + g * 16, a);
    const int kB = kk + g * 16;
#pragma unroll
    for (int j = 0; j < 16; ++j)
      b[j] = (_Float16)W[(size_t)(kB + j) * N + (tile_n + r)];
    acc = wmma16x16x32(a, b, acc);
  }
  const float bb = bias[tile_n + r];
#pragma unroll
  for (int v = 0; v < 8; ++v) {
    const int row = tile_m + v + g * 8;
    store_c(&C[(size_t)row * N + (tile_n + r)], acc[v] + bb);
  }
}

// ---------------------------------------------------------------------------
// scores = Q @ K^T / sqrt(DH) with causal mask, written into attn[B,H,S,S].
// Q,K stored f16 [B,S,D] with head h in columns [h*64, h*64+64).
// grid = (S/16/4, S/16, B*H), block = 128 (4 waves over 4 key tiles).
// ---------------------------------------------------------------------------
__global__ void scores_kernel(const _Float16* __restrict__ Q,
                              const _Float16* __restrict__ Kp,
                              float* __restrict__ attn) {
  const int lane = threadIdx.x & 31;
  const int wave = threadIdx.x >> 5;
  const int kt = blockIdx.x * 4 + wave;   // key tile
  const int qt = blockIdx.y;              // query tile
  const int bh = blockIdx.z;
  const int b  = bh / H_;
  const int h  = bh % H_;
  const int r  = lane & 15;
  const int g  = lane >> 4;

  if (kt > qt) {  // fully masked tile: exp() underflows to 0 in softmax
#pragma unroll
    for (int v = 0; v < 8; ++v) {
      const int qrow = qt * 16 + v + g * 8;
      attn[((size_t)bh * S_ + qrow) * S_ + kt * 16 + r] = -1e9f;
    }
    return;
  }

  const _Float16* Qrow = Q  + (size_t)(b * S_ + qt * 16 + r) * D_ + h * DH_;
  const _Float16* Krow = Kp + (size_t)(b * S_ + kt * 16 + r) * D_ + h * DH_;
  v8f acc = {};
#pragma unroll
  for (int kc = 0; kc < DH_; kc += 32) {
    v16h a;
    load_afrag(Qrow, kc + g * 16, a);
    // B(kdim, col=r) = K[key=kt*16+r, kdim]: 16 contiguous halves = 32-byte load
    const v16h bf = *(const v16h*)(Krow + kc + g * 16);
    acc = wmma16x16x32(a, bf, acc);
  }
#pragma unroll
  for (int v = 0; v < 8; ++v) {
    const int qrow = qt * 16 + v + g * 8;
    const int kcol = kt * 16 + r;
    float s = acc[v] * 0.125f;          // 1/sqrt(64)
    if (kcol > qrow) s = -1e9f;         // causal (matches mask*-1e9)
    attn[((size_t)bh * S_ + qrow) * S_ + kcol] = s;
  }
}

// ---------------------------------------------------------------------------
// In-place softmax over rows of length S=2048. One 256-thread block per row.
// ---------------------------------------------------------------------------
__global__ void softmax_rows_kernel(float* __restrict__ attn) {
  __shared__ float red[256];
  const int tid = threadIdx.x;
  float* row = attn + (size_t)blockIdx.x * S_;

  float vals[8];
  float m = -3.0e38f;
#pragma unroll
  for (int i = 0; i < 8; ++i) {
    vals[i] = row[tid + i * 256];
    m = fmaxf(m, vals[i]);
  }
  red[tid] = m; __syncthreads();
  for (int s = 128; s > 0; s >>= 1) {
    if (tid < s) red[tid] = fmaxf(red[tid], red[tid + s]);
    __syncthreads();
  }
  const float rowmax = red[0];
  __syncthreads();

  float ssum = 0.0f;
#pragma unroll
  for (int i = 0; i < 8; ++i) {
    vals[i] = __expf(vals[i] - rowmax);
    ssum += vals[i];
  }
  red[tid] = ssum; __syncthreads();
  for (int s = 128; s > 0; s >>= 1) {
    if (tid < s) red[tid] += red[tid + s];
    __syncthreads();
  }
  const float inv = 1.0f / red[0];
#pragma unroll
  for (int i = 0; i < 8; ++i)
    row[tid + i * 256] = vals[i] * inv;
}

// ---------------------------------------------------------------------------
// ctx[b, s, h*64 + d] = attn[b,h,s,:] @ V_h[:, d]   (head-interleaved, f16 out)
// grid = (S/16, B*H), block = 128 (4 waves over the 4 DH tiles).
// Causal structure: attn row s is zero beyond s, so K-loop stops at tile end.
// ---------------------------------------------------------------------------
__global__ void pv_kernel(const float* __restrict__ attn,
                          const _Float16* __restrict__ V,
                          _Float16* __restrict__ ctx) {
  const int lane = threadIdx.x & 31;
  const int wave = threadIdx.x >> 5;     // DH tile 0..3
  const int qt = blockIdx.x;
  const int bh = blockIdx.y;
  const int b  = bh / H_;
  const int h  = bh % H_;
  const int r  = lane & 15;
  const int g  = lane >> 4;

  const float* Arow = attn + ((size_t)bh * S_ + qt * 16 + r) * S_;
  const int ncol = h * DH_ + wave * 16 + r;

  v8f acc = {};
  const int kend = (((qt + 1) * 16) + 31) & ~31;   // past causal edge = zeros
  for (int kk = 0; kk < kend; kk += 32) {
    v16h a, bf;
    load_afrag(Arow, kk + g * 16, a);              // f32 probs -> f16
    const int kB = kk + g * 16;
#pragma unroll
    for (int j = 0; j < 16; ++j)
      bf[j] = V[(size_t)(b * S_ + kB + j) * D_ + ncol];
    acc = wmma16x16x32(a, bf, acc);
  }
#pragma unroll
  for (int v = 0; v < 8; ++v) {
    const int qrow = qt * 16 + v + g * 8;
    ctx[(size_t)(b * S_ + qrow) * D_ + ncol] = (_Float16)acc[v];
  }
}

// ---------------------------------------------------------------------------
extern "C" void kernel_launch(void* const* d_in, const int* in_sizes, int n_in,
                              void* d_out, int out_size, void* d_ws, size_t ws_size,
                              hipStream_t stream) {
  (void)in_sizes; (void)n_in; (void)out_size; (void)ws_size;

  const float* query = (const float*)d_in[0];
  const float* key   = (const float*)d_in[1];
  const float* value = (const float*)d_in[2];
  // d_in[3] is the causal mask; applied analytically in scores_kernel.
  const float* Wq = (const float*)d_in[4];
  const float* bq = (const float*)d_in[5];
  const float* Wk = (const float*)d_in[6];
  const float* bk = (const float*)d_in[7];
  const float* Wv = (const float*)d_in[8];
  const float* bv = (const float*)d_in[9];
  const float* Wo = (const float*)d_in[10];
  const float* bo = (const float*)d_in[11];

  float* out  = (float*)d_out;                      // [B,S,D]
  float* attn = out + (size_t)B_ * S_ * D_;         // [B,H,S,S]

  const size_t bsd = (size_t)B_ * S_ * D_;
  _Float16* Qb = (_Float16*)d_ws;                   // [B,S,D] f16
  _Float16* Kb = Qb + bsd;                          // [B,S,D] f16
  _Float16* Vb = Qb + 2 * bsd;                      // [B,S,D] f16
  _Float16* Cx = Qb + 3 * bsd;                      // [B,S,D] f16 context

  const int M = B_ * S_;                            // 4096
  const dim3 blk(128);
  const dim3 ggrid(D_ / 64, M / 16);

  // QKV projections (f32 activations in, f16 workspace out)
  gemm_bias_kernel<float, _Float16><<<ggrid, blk, 0, stream>>>(query, Wq, bq, Qb, M, D_, D_);
  gemm_bias_kernel<float, _Float16><<<ggrid, blk, 0, stream>>>(key,   Wk, bk, Kb, M, D_, D_);
  gemm_bias_kernel<float, _Float16><<<ggrid, blk, 0, stream>>>(value, Wv, bv, Vb, M, D_, D_);

  // scores (with causal mask) -> attn region (f32 output)
  scores_kernel<<<dim3(S_ / 16 / 4, S_ / 16, B_ * H_), blk, 0, stream>>>(Qb, Kb, attn);

  // softmax in place over each of B*H*S rows
  softmax_rows_kernel<<<dim3(B_ * H_ * S_), dim3(256), 0, stream>>>(attn);

  // context = attn @ V (head-interleaved layout, f16 out)
  pv_kernel<<<dim3(S_ / 16, B_ * H_), blk, 0, stream>>>(attn, Vb, Cx);

  // output projection (f16 ctx in, f32 out)
  gemm_bias_kernel<_Float16, float><<<ggrid, blk, 0, stream>>>(Cx, Wo, bo, out, M, D_, D_);
}
